// RecurrentGNN_90503550861446
// MI455X (gfx1250) — matrix operand order repaired and checked by reading, not compiled, not run on hardware
//
#include <hip/hip_runtime.h>
#include <hip/hip_bf16.h>

// ---------------------------------------------------------------------------
// Recurrent GNN on MI455X (gfx1250), bf16 WMMA with f32 accumulation.
//   N=65536 nodes, DH=32, DM=64, L=2 layers, E=2097152 edges, T=64 steps
// ---------------------------------------------------------------------------

#define NNODES 65536
#define DHF    32
#define DMF    64
#define NEDGES 2097152
#define TSTEPS 64
#define TPW    8          // edge tiles per wave (amortizes weight/LDS staging)

typedef __attribute__((ext_vector_type(16))) __bf16 v16bf;
typedef __attribute__((ext_vector_type(8)))  float  v8f;
typedef __attribute__((ext_vector_type(4)))  float  fvec4;
typedef __attribute__((ext_vector_type(4)))  int    i4vec;

union AFrag { v16bf v; __bf16 e[16]; };

#if defined(__has_builtin)
#if __has_builtin(__builtin_amdgcn_global_load_async_to_lds_b128)
#define HAVE_ASYNC_LDS 1
#endif
#if __has_builtin(__builtin_amdgcn_s_wait_asynccnt)
#define HAVE_ASYNC_WAIT_BUILTIN 1
#endif
#endif

#if defined(HAVE_ASYNC_LDS)
typedef __attribute__((address_space(1))) i4vec* gas_p;   // global int4*
typedef __attribute__((address_space(3))) i4vec* las_p;   // LDS int4*
__device__ __forceinline__ void async_wait0() {
#if defined(HAVE_ASYNC_WAIT_BUILTIN)
  __builtin_amdgcn_s_wait_asynccnt(0);
#else
  asm volatile("s_wait_asynccnt 0x0" ::: "memory");
#endif
}
#endif

// copy one 64-byte bf16 row (global -> LDS), async path when available
__device__ __forceinline__ void gather_row64(const __bf16* gsrc, __bf16* ldst) {
#if defined(HAVE_ASYNC_LDS)
  gas_p g = (gas_p)gsrc;
  las_p l = (las_p)ldst;
#pragma unroll
  for (int k = 0; k < 4; ++k)
    __builtin_amdgcn_global_load_async_to_lds_b128(g + k, l + k, 0, 0);
#else
  const uint4* g = (const uint4*)gsrc;
  uint4* l = (uint4*)ldst;
#pragma unroll
  for (int k = 0; k < 4; ++k) l[k] = g[k];
#endif
}

__device__ __forceinline__ void gather_fence() {
#if defined(HAVE_ASYNC_LDS)
  async_wait0();
#endif
}

// --- A fragment (16x32 bf16, MxK) from a row-major LDS tile ---------------
// lane L: M = L&15, kb = (L<16)?0:8; elem j<8 -> K=kb+j ; j>=8 -> K=kb+8+j
__device__ __forceinline__ v16bf load_A_frag(const __bf16* tile, int stride,
                                             int k0, int lane) {
  int M  = lane & 15;
  int kb = (lane < 16) ? 0 : 8;
  const __bf16* p = tile + M * stride + k0 + kb;
  AFrag a;
#pragma unroll
  for (int j = 0; j < 8; ++j) { a.e[j] = p[j]; a.e[j + 8] = p[16 + j]; }
  return a.v;
}

// --- B fragment (32x16 bf16, KxN): [lane][16] -----------------------------
__device__ __forceinline__ v16bf load_B_frag(const __bf16* frag, int lane) {
  return ((const v16bf*)frag)[lane];
}

// silu with fast hardware reciprocal (result is rounded to bf16 anyway)
__device__ __forceinline__ float silu(float x) {
  return x * __builtin_amdgcn_rcpf(1.0f + __expf(-x));
}

__device__ __forceinline__ void atomicAddF32(float* p, float v) {
  __hip_atomic_fetch_add(p, v, __ATOMIC_RELAXED, __HIP_MEMORY_SCOPE_AGENT);
}

// ---------------------------------------------------------------------------
// Weight packing:  48 fragments of 512 bf16 ([32 lanes][16 elems]).
// Per layer l (24 frags): Wm1 kt{0,1}xnt{0..3} -> 0..7 ; Wm2 -> 8..15 ;
//                         Wu1 kt{0..2}xnt{0,1} -> 16..21 ; Wu2 nt{0,1} -> 22..23
// ---------------------------------------------------------------------------
__global__ __launch_bounds__(512) void pack_weights_kernel(
    const float* __restrict__ Wm1, const float* __restrict__ Wm2,
    const float* __restrict__ Wu1, const float* __restrict__ Wu2,
    __bf16* __restrict__ pw) {
  int b    = blockIdx.x;       // 0..47
  int tid  = threadIdx.x;      // 0..511
  int lane = tid >> 4;
  int j    = tid & 15;
  int l    = b / 24;
  int r    = b % 24;
  const float* src;
  int Ndim, kt, nt;
  if (r < 8)        { kt = r >> 2;        nt = r & 3;        src = Wm1 + l * 64 * 64; Ndim = 64; }
  else if (r < 16)  { int q = r - 8;  kt = q >> 2; nt = q & 3; src = Wm2 + l * 64 * 64; Ndim = 64; }
  else if (r < 22)  { int q = r - 16; kt = q >> 1; nt = q & 1; src = Wu1 + l * 96 * 32; Ndim = 32; }
  else              { kt = 0;             nt = r - 22;        src = Wu2 + l * 32 * 32; Ndim = 32; }
  int K = kt * 32 + ((lane < 16) ? 0 : 8) + ((j < 8) ? j : 8 + j);  // B layout: table order (matches A)
  int N = nt * 16 + (lane & 15);
  pw[b * 512 + tid] = (__bf16)src[K * Ndim + N];
}

// ---------------------------------------------------------------------------
__global__ __launch_bounds__(256) void init_kernel(const float* __restrict__ h0,
                                                   float* __restrict__ h_cur,
                                                   __bf16* __restrict__ hb_cur,
                                                   float* __restrict__ agg) {
  int i = blockIdx.x * 256 + threadIdx.x;          // covers N*DM
  agg[i] = 0.0f;
  if (i < NNODES * DHF) {
    float v = h0[i];
    h_cur[i]  = v;
    hb_cur[i] = (__bf16)v;
  }
}

// ---------------------------------------------------------------------------
// Edge MLP + scatter.  Wave = 16 edges/tile, TPW tiles/wave, 8 waves/block.
//   x = [hb[src], hb[dst]] (16x64) -> silu(x@Wm1+b) -> @Wm2+b
//   -> atomicAdd agg[dst].   Wm fragments staged once per block in LDS.
// ---------------------------------------------------------------------------
__global__ __launch_bounds__(256) void edge_msg_kernel(
    const __bf16* __restrict__ hb, const int* __restrict__ edge_index,
    const __bf16* __restrict__ pw, const float* __restrict__ bm1,
    const float* __restrict__ bm2, float* __restrict__ agg, int layer) {
  __shared__ __align__(16) __bf16 Wf[16 * 512];      // Wm1 frags 0..7, Wm2 8..15
  __shared__ __align__(16) __bf16 Xs[8][16][64];
  __shared__ __align__(16) __bf16 Ys[8][16][64];
  __shared__ int Ds[8][16];

  int tid = threadIdx.x, wave = tid >> 5, lane = tid & 31;

  // ---- stage this layer's Wm fragments (16KB) cooperatively --------------
  {
    const uint4* src = (const uint4*)(pw + (size_t)layer * 24 * 512);
    uint4* dst = (uint4*)Wf;
#pragma unroll
    for (int i = 0; i < 4; ++i) dst[tid + 256 * i] = src[tid + 256 * i];
  }
  int Nloc  = lane & 15;
  int Mbase = (lane < 16) ? 0 : 8;
  float bias1[4], bias2[4];
#pragma unroll
  for (int nt = 0; nt < 4; ++nt) {
    bias1[nt] = bm1[layer * DMF + nt * 16 + Nloc];
    bias2[nt] = bm2[layer * DMF + nt * 16 + Nloc];
  }
  __syncthreads();

  int m = lane & 15, half = lane >> 4;
  long waveBase = ((long)blockIdx.x * 8 + wave) * (TPW * 16);

  for (int tIt = 0; tIt < TPW; ++tIt) {
    long tileBase = waveBase + (long)tIt * 16;

    // ---- gather: lane = (edge row m, half=src/dst) -> 64B bf16 row -------
    int idx = edge_index[(long)half * NEDGES + tileBase + m];
    if (half) Ds[wave][m] = idx;
    gather_row64(hb + (size_t)idx * DHF, &Xs[wave][m][half * 32]);
    gather_fence();

    // ---- GEMM1: (16x64)@(64x64), silu -> Ys ------------------------------
    v16bf a0 = load_A_frag(&Xs[wave][0][0], 64, 0, lane);
    v16bf a1 = load_A_frag(&Xs[wave][0][0], 64, 32, lane);
#pragma unroll
    for (int nt = 0; nt < 4; ++nt) {
      v8f acc = {};
      v16bf b0 = load_B_frag(Wf + (0 * 4 + nt) * 512, lane);
      v16bf b1 = load_B_frag(Wf + (1 * 4 + nt) * 512, lane);
      acc = __builtin_amdgcn_wmma_f32_16x16x32_bf16(false, a0, false, b0, (short)0, acc, false, false);
      acc = __builtin_amdgcn_wmma_f32_16x16x32_bf16(false, a1, false, b1, (short)0, acc, false, false);
#pragma unroll
      for (int r = 0; r < 8; ++r)
        Ys[wave][Mbase + r][nt * 16 + Nloc] = (__bf16)silu(acc[r] + bias1[nt]);
    }

    // ---- GEMM2: (16x64)@(64x64) -> scatter atomics -----------------------
    v16bf c0 = load_A_frag(&Ys[wave][0][0], 64, 0, lane);
    v16bf c1 = load_A_frag(&Ys[wave][0][0], 64, 32, lane);
    int dsts[8];
#pragma unroll
    for (int r = 0; r < 8; ++r) dsts[r] = Ds[wave][Mbase + r];
#pragma unroll
    for (int nt = 0; nt < 4; ++nt) {
      v8f acc = {};
      v16bf b0 = load_B_frag(Wf + (8 + 0 * 4 + nt) * 512, lane);
      v16bf b1 = load_B_frag(Wf + (8 + 1 * 4 + nt) * 512, lane);
      acc = __builtin_amdgcn_wmma_f32_16x16x32_bf16(false, c0, false, b0, (short)0, acc, false, false);
      acc = __builtin_amdgcn_wmma_f32_16x16x32_bf16(false, c1, false, b1, (short)0, acc, false, false);
#pragma unroll
      for (int r = 0; r < 8; ++r)
        atomicAddF32(&agg[(long)dsts[r] * DMF + nt * 16 + Nloc], acc[r] + bias2[nt]);
    }
  }
}

// ---------------------------------------------------------------------------
// Node MLP (+ residual; last layer also Euler step + readout).
//   u = [h, agg] (16x96) -> silu(u@Wu1+b) -> @Wu2+b ; out = h_in + mlp
//   last: h_new = h_prev + dt*(out - h_prev);  readout = h_new . w_ro + b_ro
// Zeroes agg after consuming it; maintains a bf16 mirror of its output.
// ---------------------------------------------------------------------------
__global__ __launch_bounds__(256) void node_update_kernel(
    const float* __restrict__ h_in, const __bf16* __restrict__ hb_in,
    const float* __restrict__ h_prev, float* __restrict__ h_out,
    __bf16* __restrict__ hb_out, float* __restrict__ agg,
    const __bf16* __restrict__ pw, const float* __restrict__ bu1,
    const float* __restrict__ bu2, const float* __restrict__ w_ro,
    const float* __restrict__ b_ro, const float* __restrict__ tvec,
    float* __restrict__ out, int layer, int last, int ti) {
  __shared__ __align__(16) __bf16 Wf[8 * 512];       // Wu1 frags 0..5, Wu2 6..7
  __shared__ __align__(16) __bf16 Us[8][16][96];
  __shared__ __align__(16) __bf16 Ys[8][16][32];
  __shared__ float Hn[8][16][32];

  int tid = threadIdx.x, wave = tid >> 5, lane = tid & 31;

  // ---- stage Wu fragments (8KB) cooperatively ----------------------------
  {
    const uint4* src = (const uint4*)(pw + ((size_t)layer * 24 + 16) * 512);
    uint4* dst = (uint4*)Wf;
#pragma unroll
    for (int i = 0; i < 2; ++i) dst[tid + 256 * i] = src[tid + 256 * i];
  }
  int Nloc  = lane & 15;
  int Mbase = (lane < 16) ? 0 : 8;
  float biasU1[2], biasU2[2];
#pragma unroll
  for (int nt = 0; nt < 2; ++nt) {
    biasU1[nt] = bu1[layer * DHF + nt * 16 + Nloc];
    biasU2[nt] = bu2[layer * DHF + nt * 16 + Nloc];
  }

  int tile = blockIdx.x * 8 + wave;
  int m    = lane & 15;
  int half = lane >> 4;
  int node = tile * 16 + m;

  // ---- gather u = [hb(32) | agg(64)] into LDS ----------------------------
  if (half == 0) {
    const uint4* g = (const uint4*)(hb_in + (size_t)node * DHF);   // 64B bf16
    uint4* l = (uint4*)&Us[wave][m][0];
#pragma unroll
    for (int k = 0; k < 4; ++k) l[k] = g[k];
  } else {
    const float* ap = agg + (long)node * DMF;
#pragma unroll
    for (int k4 = 0; k4 < 16; ++k4) {
      fvec4 v = ((const fvec4*)ap)[k4];
      __bf16* u = &Us[wave][m][32 + k4 * 4];
      u[0] = (__bf16)v.x; u[1] = (__bf16)v.y; u[2] = (__bf16)v.z; u[3] = (__bf16)v.w;
    }
  }
  __syncthreads();

  // zero agg for the next accumulation pass (reads are done)
  if (half == 1) {
    fvec4 z = {};
    float* ap = agg + (long)node * DMF;
#pragma unroll
    for (int k4 = 0; k4 < 16; ++k4) ((fvec4*)ap)[k4] = z;
  }

  // ---- GEMM1: (16x96)@(96x32), silu -> Ys --------------------------------
  v16bf a0 = load_A_frag(&Us[wave][0][0], 96, 0, lane);
  v16bf a1 = load_A_frag(&Us[wave][0][0], 96, 32, lane);
  v16bf a2 = load_A_frag(&Us[wave][0][0], 96, 64, lane);
#pragma unroll
  for (int nt = 0; nt < 2; ++nt) {
    v8f acc = {};
    v16bf b0 = load_B_frag(Wf + (0 * 2 + nt) * 512, lane);
    v16bf b1 = load_B_frag(Wf + (1 * 2 + nt) * 512, lane);
    v16bf b2 = load_B_frag(Wf + (2 * 2 + nt) * 512, lane);
    acc = __builtin_amdgcn_wmma_f32_16x16x32_bf16(false, a0, false, b0, (short)0, acc, false, false);
    acc = __builtin_amdgcn_wmma_f32_16x16x32_bf16(false, a1, false, b1, (short)0, acc, false, false);
    acc = __builtin_amdgcn_wmma_f32_16x16x32_bf16(false, a2, false, b2, (short)0, acc, false, false);
#pragma unroll
    for (int r = 0; r < 8; ++r)
      Ys[wave][Mbase + r][nt * 16 + Nloc] = (__bf16)silu(acc[r] + biasU1[nt]);
  }

  // ---- GEMM2: (16x32)@(32x32) + residual / Euler -------------------------
  v16bf c0 = load_A_frag(&Ys[wave][0][0], 32, 0, lane);
  float dt = 0.0f;
  if (last && ti > 0) dt = tvec[ti] - tvec[ti - 1];
#pragma unroll
  for (int nt = 0; nt < 2; ++nt) {
    v8f acc = {};
    v16bf b0 = load_B_frag(Wf + (6 + nt) * 512, lane);
    acc = __builtin_amdgcn_wmma_f32_16x16x32_bf16(false, c0, false, b0, (short)0, acc, false, false);
#pragma unroll
    for (int r = 0; r < 8; ++r) {
      int nrow = tile * 16 + Mbase + r;
      int f    = nt * 16 + Nloc;
      float hin = h_in[(long)nrow * DHF + f];   // exact f32 residual
      float val = hin + acc[r] + biasU2[nt];    // out_{l+1}
      if (last) {
        float hp0  = h_prev[(long)nrow * DHF + f];
        float hnew = hp0 + dt * (val - hp0);
        h_out[(long)nrow * DHF + f]  = hnew;
        hb_out[(long)nrow * DHF + f] = (__bf16)hnew;
        Hn[wave][Mbase + r][f] = hnew;
      } else {
        h_out[(long)nrow * DHF + f]  = val;
        hb_out[(long)nrow * DHF + f] = (__bf16)val;
      }
    }
  }

  // readout: Hn is wave-private; same-wave LDS ordering suffices
  if (last && lane < 16) {
    float s = b_ro[0];
    const float* hn = &Hn[wave][lane][0];
#pragma unroll
    for (int f = 0; f < DHF; ++f) s += hn[f] * w_ro[f];
    out[(long)ti * NNODES + tile * 16 + lane] = s;
  }
}

// ---------------------------------------------------------------------------
extern "C" void kernel_launch(void* const* d_in, const int* in_sizes, int n_in,
                              void* d_out, int out_size, void* d_ws, size_t ws_size,
                              hipStream_t stream) {
  const float* t    = (const float*)d_in[0];
  const int*   eidx = (const int*)d_in[1];
  const float* h0   = (const float*)d_in[2];
  const float* Wm1  = (const float*)d_in[3];
  const float* bm1  = (const float*)d_in[4];
  const float* Wm2  = (const float*)d_in[5];
  const float* bm2  = (const float*)d_in[6];
  const float* Wu1  = (const float*)d_in[7];
  const float* bu1  = (const float*)d_in[8];
  const float* Wu2  = (const float*)d_in[9];
  const float* bu2  = (const float*)d_in[10];
  const float* w_ro = (const float*)d_in[11];
  const float* b_ro = (const float*)d_in[12];
  float* out = (float*)d_out;

  // ws: h_cur 8M | h_mid 8M | agg 16M | hb_cur 4M | hb_mid 4M | packed W 48K
  char*   ws     = (char*)d_ws;
  float*  h_cur  = (float*)(ws);
  float*  h_mid  = (float*)(ws + (size_t)8388608);
  float*  agg    = (float*)(ws + (size_t)16777216);
  __bf16* hb_cur = (__bf16*)(ws + (size_t)33554432);
  __bf16* hb_mid = (__bf16*)(ws + (size_t)37748736);
  __bf16* pw     = (__bf16*)(ws + (size_t)41943040);

  pack_weights_kernel<<<48, 512, 0, stream>>>(Wm1, Wm2, Wu1, Wu2, pw);
  init_kernel<<<(NNODES * DMF) / 256, 256, 0, stream>>>(h0, h_cur, hb_cur, agg);

  for (int ti = 0; ti < TSTEPS; ++ti) {
    // layer 0: reads h_cur/hb_cur, writes h_mid/hb_mid
    edge_msg_kernel<<<NEDGES / (128 * TPW), 256, 0, stream>>>(hb_cur, eidx, pw, bm1, bm2, agg, 0);
    node_update_kernel<<<NNODES / 128, 256, 0, stream>>>(h_cur, hb_cur, h_cur, h_mid, hb_mid,
                                                         agg, pw, bu1, bu2, w_ro, b_ro, t, out,
                                                         0, 0, ti);
    // layer 1: reads h_mid/hb_mid, Euler-integrates into h_cur, readout row ti
    edge_msg_kernel<<<NEDGES / (128 * TPW), 256, 0, stream>>>(hb_mid, eidx, pw, bm1, bm2, agg, 1);
    node_update_kernel<<<NNODES / 128, 256, 0, stream>>>(h_mid, hb_mid, h_cur, h_cur, hb_cur,
                                                         agg, pw, bu1, bu2, w_ro, b_ro, t, out,
                                                         1, 1, ti);
  }
}
